// token_selection_13477607375006
// MI455X (gfx1250) — matrix operand order repaired and checked by reading, not compiled, not run on hardware
//
#include <hip/hip_runtime.h>

// Problem constants (match reference)
#define BB   32
#define NN   4096
#define DD   256
#define DH   128
#define KTOP 2048          // ceil(N * 0.5)
#define MTOT (BB * NN)     // 131072 rows

typedef __attribute__((ext_vector_type(16))) __bf16 v16bf;
typedef __attribute__((ext_vector_type(8)))  float  v8f;
typedef __attribute__((ext_vector_type(4)))  int    v4i;

union Frag16 {
    v16bf v;
    uint4 q[2];
};

// ---- CDNA5 async global->LDS copy (ASYNCcnt path), with safe fallbacks ----
#if __has_builtin(__builtin_amdgcn_global_load_async_to_lds_b128)
#define HAVE_ASYNC_LDS 1
#else
#define HAVE_ASYNC_LDS 0
#endif

typedef __attribute__((address_space(1))) v4i gv4i;   // global int4
typedef __attribute__((address_space(3))) v4i lv4i;   // LDS int4

__device__ inline void async_cp16(const void* g, void* l) {
#if HAVE_ASYNC_LDS
    __builtin_amdgcn_global_load_async_to_lds_b128((gv4i*)g, (lv4i*)l, 0, 0);
#else
    *(uint4*)l = *(const uint4*)g;
#endif
}

__device__ inline void wait_async0() {
#if HAVE_ASYNC_LDS
#if __has_builtin(__builtin_amdgcn_s_wait_asynccnt)
    __builtin_amdgcn_s_wait_asynccnt(0);
#else
    asm volatile("s_wait_asynccnt 0x0" ::: "memory");
#endif
#endif
}

// ---------------------------------------------------------------------------
// Kernel -1: one-time weight conversion f32 -> bf16 (removes recurring f32
// weight traffic: Wf was re-read 1024x, Ws 2048x by the GEMM grids).
// ---------------------------------------------------------------------------
__global__ __launch_bounds__(256)
void convert_weights_kernel(const float* __restrict__ Wf, const float* __restrict__ Ws,
                            __bf16* __restrict__ WfB, __bf16* __restrict__ WsB) {
    int i = blockIdx.x * 256 + threadIdx.x;        // grid covers 65536
    if (i < DH * DD) WfB[i] = (__bf16)Wf[i];
    if (i < DD * DD) WsB[i] = (__bf16)Ws[i];
}

// ---------------------------------------------------------------------------
// Kernel 0: half_cls = cls @ W_f^T + b_f   (NO ReLU on this branch)
// ---------------------------------------------------------------------------
__global__ __launch_bounds__(128)
void cls_mlp_kernel(const float* __restrict__ cls, const float* __restrict__ Wf,
                    const float* __restrict__ bf, __bf16* __restrict__ halfCls) {
    const int b = blockIdx.x;         // 0..31
    const int h = threadIdx.x;        // 0..127
    const float* c = cls + (size_t)b * DD;
    const float* w = Wf + (size_t)h * DD;
    float s = 0.f;
#pragma unroll 8
    for (int d = 0; d < DD; ++d) s += c[d] * w[d];
    s += bf[h];
    halfCls[(size_t)b * DH + h] = (__bf16)s;
}

// ---------------------------------------------------------------------------
// Kernel 1: half_patch = ReLU(X @ W_f^T + b_f), bf16 out [MTOT][128]
// 256 threads (8 waves). Tile 128M x 128H, K chunks of 32, double-buffered
// LDS; B tile staged via async global->LDS (pure bf16 copy).
// ---------------------------------------------------------------------------
__global__ __launch_bounds__(256)
void gemm1_kernel(const float* __restrict__ X, const __bf16* __restrict__ WfB,
                  const float* __restrict__ bf, __bf16* __restrict__ halfPatch) {
    __shared__ __align__(16) __bf16 As[2][128][32];   // 16 KB
    __shared__ __align__(16) __bf16 Bs[2][128][32];   // 16 KB

    const int tid  = threadIdx.x;
    const int wave = tid >> 5;
    const int lane = tid & 31;
    const size_t m0 = (size_t)blockIdx.x * 128;
    const int mw = wave * 16;

    v8f acc[8] = {};

    auto stage = [&](int k0, int buf) {
        // A: X f32 -> bf16 convert (X is read exactly once; convert in flight)
#pragma unroll
        for (int i = 0; i < 4; ++i) {
            int id = tid + i * 256;
            int r = id >> 3, c4 = (id & 7) << 2;
            float4 v = *(const float4*)(X + (m0 + r) * (size_t)DD + k0 + c4);
            As[buf][r][c4 + 0] = (__bf16)v.x;
            As[buf][r][c4 + 1] = (__bf16)v.y;
            As[buf][r][c4 + 2] = (__bf16)v.z;
            As[buf][r][c4 + 3] = (__bf16)v.w;
        }
        // B: pure bf16 copy -> async to LDS
#pragma unroll
        for (int i = 0; i < 2; ++i) {
            int id = tid + i * 256;
            int r = id >> 2, cc = (id & 3) << 3;
            async_cp16(WfB + (size_t)r * DD + k0 + cc, &Bs[buf][r][cc]);
        }
    };

    stage(0, 0);
    wait_async0();
    __syncthreads();

#pragma unroll
    for (int ki = 0; ki < 8; ++ki) {
        const int p = ki & 1;
        if (ki < 7) stage((ki + 1) * 32, p ^ 1);   // overlap copy with WMMA

        Frag16 a;
        const int am = mw + (lane & 15);
        const int kh = (lane >> 4) * 16;
        a.q[0] = *(const uint4*)&As[p][am][kh];
        a.q[1] = *(const uint4*)&As[p][am][kh + 8];

#pragma unroll
        for (int h = 0; h < 8; ++h) {
            Frag16 bfr;
            const int hn = h * 16 + (lane & 15);
            const int kb = (lane >> 4) * 16;
            bfr.q[0] = *(const uint4*)&Bs[p][hn][kb];
            bfr.q[1] = *(const uint4*)&Bs[p][hn][kb + 8];
            acc[h] = __builtin_amdgcn_wmma_f32_16x16x32_bf16(
                false, a.v, false, bfr.v, (short)0, acc[h], false, false);
        }
        if (ki < 7) wait_async0();
        __syncthreads();
    }

    const int n  = lane & 15;
    const int rb = (lane >> 4) * 8;
#pragma unroll
    for (int h = 0; h < 8; ++h) {
        const int col  = h * 16 + n;
        const float bias = bf[col];
#pragma unroll
        for (int v = 0; v < 8; ++v) {
            float val = acc[h][v] + bias;
            val = val > 0.f ? val : 0.f;
            size_t m = m0 + mw + rb + v;
            halfPatch[m * DH + col] = (__bf16)val;
        }
    }
}

// ---------------------------------------------------------------------------
// Kernel 2: logits = ReLU(concat(half_cls, half_patch) @ W_s^T + b_s), f32
// grid.x = (MTOT/128)*2 ; block parity selects E-half. A upper-half and B
// tiles are pure bf16 copies -> async global->LDS, double-buffered.
// ---------------------------------------------------------------------------
__global__ __launch_bounds__(256)
void gemm2_kernel(const __bf16* __restrict__ halfPatch,
                  const __bf16* __restrict__ halfCls,
                  const __bf16* __restrict__ WsB, const float* __restrict__ bs,
                  float* __restrict__ logits) {
    __shared__ __align__(16) __bf16 As[2][128][32];
    __shared__ __align__(16) __bf16 Bs[2][128][32];

    const int tid  = threadIdx.x;
    const int wave = tid >> 5;
    const int lane = tid & 31;
    const int eblk = blockIdx.x & 1;
    const size_t m0 = (size_t)(blockIdx.x >> 1) * 128;
    const int e0 = eblk * 128;
    const int b  = (int)(m0 >> 12);
    const int mw = wave * 16;

    v8f acc[8] = {};

    auto stage = [&](int k0, int buf) {
        if (k0 < DH) {
            // concat cols 0..127 = half_cls[b] broadcast to all 128 rows
            const int col = tid & 31;
            __bf16 cv = halfCls[(size_t)b * DH + k0 + col];
            for (int r = tid >> 5; r < 128; r += 8) As[buf][r][col] = cv;
        } else {
            // concat cols 128..255 = half_patch rows: async bf16 copy
#pragma unroll
            for (int i = 0; i < 2; ++i) {
                int id = tid + i * 256;
                int r = id >> 2, cc = (id & 3) << 3;
                async_cp16(halfPatch + (m0 + r) * DH + (k0 - DH) + cc,
                           &As[buf][r][cc]);
            }
        }
#pragma unroll
        for (int i = 0; i < 2; ++i) {
            int id = tid + i * 256;
            int r = id >> 2, cc = (id & 3) << 3;
            async_cp16(WsB + (size_t)(e0 + r) * DD + k0 + cc, &Bs[buf][r][cc]);
        }
    };

    stage(0, 0);
    wait_async0();
    __syncthreads();

#pragma unroll
    for (int ki = 0; ki < 8; ++ki) {
        const int p = ki & 1;
        if (ki < 7) stage((ki + 1) * 32, p ^ 1);

        Frag16 a;
        const int am = mw + (lane & 15);
        const int kh = (lane >> 4) * 16;
        a.q[0] = *(const uint4*)&As[p][am][kh];
        a.q[1] = *(const uint4*)&As[p][am][kh + 8];

#pragma unroll
        for (int h = 0; h < 8; ++h) {
            Frag16 bfr;
            const int hn = h * 16 + (lane & 15);
            const int kb = (lane >> 4) * 16;
            bfr.q[0] = *(const uint4*)&Bs[p][hn][kb];
            bfr.q[1] = *(const uint4*)&Bs[p][hn][kb + 8];
            acc[h] = __builtin_amdgcn_wmma_f32_16x16x32_bf16(
                false, a.v, false, bfr.v, (short)0, acc[h], false, false);
        }
        if (ki < 7) wait_async0();
        __syncthreads();
    }

    const int n  = lane & 15;
    const int rb = (lane >> 4) * 8;
#pragma unroll
    for (int h = 0; h < 8; ++h) {
        const int col  = e0 + h * 16 + n;
        const float bias = bs[col];
#pragma unroll
        for (int v = 0; v < 8; ++v) {
            float val = acc[h][v] + bias;
            val = val > 0.f ? val : 0.f;
            size_t m = m0 + mw + rb + v;
            logits[m * DD + col] = val;
        }
    }
}

// ---------------------------------------------------------------------------
// Kernel 3: per-(b,channel) top-k via full bitonic sort of 4096 keys in LDS.
// Softmax is monotone per column -> top-k on post-ReLU logits is identical.
// Key = (floatBits << 32) | (4095 - token): value desc, token asc tie-break.
// ---------------------------------------------------------------------------
__global__ __launch_bounds__(512)
void topk_sort_kernel(const float* __restrict__ logits, int* __restrict__ idxT) {
    __shared__ unsigned long long keys[NN];
    const int tid = threadIdx.x;
    const int b = blockIdx.x >> 8;
    const int c = blockIdx.x & 255;

    for (int i = tid; i < NN; i += 512) {
        float v = logits[((size_t)b * NN + i) * DD + c];   // v >= 0 after ReLU
        unsigned u = __float_as_uint(v);
        keys[i] = ((unsigned long long)u << 32) | (unsigned)(NN - 1 - i);
    }
    __syncthreads();

    for (int k = 2; k <= NN; k <<= 1) {
        for (int j = k >> 1; j > 0; j >>= 1) {
            for (int i = tid; i < NN; i += 512) {
                int ixj = i ^ j;
                if (ixj > i) {
                    unsigned long long a = keys[i], bb = keys[ixj];
                    bool descSeg = ((i & k) == 0);
                    if (descSeg ? (a < bb) : (a > bb)) { keys[i] = bb; keys[ixj] = a; }
                }
            }
            __syncthreads();
        }
    }

    int* outp = idxT + (size_t)blockIdx.x * KTOP;   // [b][c][j] contiguous
    for (int i = tid; i < KTOP; i += 512)
        outp[i] = (NN - 1) - (int)(keys[i] & 0xFFFFFFFFu);
}

// ---------------------------------------------------------------------------
// Kernel 4: gather  out[b][j][c] = X[b][idx[b][c][j]][c]
// Coalesced output; scattered reads stay in the 192 MB L2 (X slice = 128 MB).
// ---------------------------------------------------------------------------
__global__ __launch_bounds__(256)
void gather_kernel(const float* __restrict__ X, const int* __restrict__ idxT,
                   float* __restrict__ out) {
    const int c  = threadIdx.x;
    const int b  = blockIdx.x >> 7;
    const int j0 = (blockIdx.x & 127) * 16;

    int4 iv[4];
    const int4* ip = (const int4*)(idxT + ((size_t)(b * DD + c)) * KTOP + j0);
#pragma unroll
    for (int q = 0; q < 4; ++q) iv[q] = ip[q];
    const int* idx = (const int*)iv;

    const float* xb = X + (size_t)b * NN * DD;
    float* ob = out + ((size_t)b * KTOP + j0) * DD;
#pragma unroll
    for (int jj = 0; jj < 16; ++jj) {
        float v = xb[(size_t)idx[jj] * DD + c];
        ob[(size_t)jj * DD + c] = v;
    }
}

// ---------------------------------------------------------------------------
extern "C" void kernel_launch(void* const* d_in, const int* in_sizes, int n_in,
                              void* d_out, int out_size, void* d_ws, size_t ws_size,
                              hipStream_t stream) {
    const float* X   = (const float*)d_in[0];   // [32,4096,256]
    const float* cls = (const float*)d_in[1];   // [32,256]
    const float* Wf  = (const float*)d_in[2];   // [128,256]
    const float* bf  = (const float*)d_in[3];   // [128]
    const float* Ws  = (const float*)d_in[4];   // [256,256]
    const float* bs  = (const float*)d_in[5];   // [256]
    float* out = (float*)d_out;                 // [32,2048,256]

    char* ws = (char*)d_ws;
    const size_t o_wfb = 0;                                   // 64 KB bf16
    const size_t o_wsb = o_wfb + (size_t)DH * DD * 2;         // 128 KB bf16
    const size_t o_cls = o_wsb + (size_t)DD * DD * 2;         // 8 KB bf16
    const size_t o_hp  = o_cls + (size_t)BB * DH * 2;         // 32 MB bf16
    const size_t o_log = o_hp + (size_t)MTOT * DH * 2;        // 128 MB f32
    const size_t o_idx = o_log + (size_t)MTOT * DD * 4;       // 64 MB i32

    __bf16* WfB       = (__bf16*)(ws + o_wfb);
    __bf16* WsB       = (__bf16*)(ws + o_wsb);
    __bf16* halfCls   = (__bf16*)(ws + o_cls);
    __bf16* halfPatch = (__bf16*)(ws + o_hp);
    float*  logits    = (float*)(ws + o_log);
    int*    idxT      = (int*)(ws + o_idx);

    convert_weights_kernel<<<(DD * DD) / 256, 256, 0, stream>>>(Wf, Ws, WfB, WsB);
    cls_mlp_kernel<<<BB, DH, 0, stream>>>(cls, Wf, bf, halfCls);
    gemm1_kernel<<<MTOT / 128, 256, 0, stream>>>(X, WfB, bf, halfPatch);
    gemm2_kernel<<<(MTOT / 128) * 2, 256, 0, stream>>>(halfPatch, halfCls, WsB, bs, logits);
    topk_sort_kernel<<<BB * DD, 512, 0, stream>>>(logits, idxT);
    gather_kernel<<<BB * (KTOP / 16), 256, 0, stream>>>(X, idxT, out);
}
